// SFR_26834955665627
// MI455X (gfx1250) — matrix operand order repaired and checked by reading, not compiled, not run on hardware
//
#include <hip/hip_runtime.h>
#include <hip/hip_bf16.h>
#include <math.h>

typedef __attribute__((ext_vector_type(16))) _Float16 v16h;
typedef __attribute__((ext_vector_type(8)))  _Float16 v8h;
typedef __attribute__((ext_vector_type(8)))  float    v8f;
typedef __attribute__((ext_vector_type(4)))  float    nf4;

// ---------------------------------------------------------------------------
// Problem constants
// ---------------------------------------------------------------------------
#define BB   16
#define CB   256
#define CS   512
#define HH   128
#define WW   128
#define HID  512
#define H2   64
#define W2   64
#define TOKH 1024   // 32x32 tokens per batch for high bands (PH=2 pooling)
#define TOKL 256    // 16x16 tokens per batch for low path (PL=4 pooling)
#define MH   (BB*TOKH)  // 16384 token rows per band
#define ML   (BB*TOKL)  // 4096 token rows, low path

// ---------------------------------------------------------------------------
// Non-temporal streaming helpers (TH=NT: keep 268MB fp32 streams out of L2,
// preserving L2 residency for tokens/hidden/weights that WMMA GEMMs reuse)
// ---------------------------------------------------------------------------
__device__ __forceinline__ nf4 nt_load4(const float* p) {
    return __builtin_nontemporal_load((const nf4*)p);
}
__device__ __forceinline__ void nt_store4(float* p, nf4 v) {
    __builtin_nontemporal_store(v, (nf4*)p);
}

// ---------------------------------------------------------------------------
// WMMA B-fragment loader (from pre-transposed NxK weight): lane&15 = N column,
// (lane>>4)*16 = K base, 16 contiguous K halves per lane (this file's k order, same as A).
// ---------------------------------------------------------------------------
__device__ __forceinline__ v16h load_b_frag(const _Float16* __restrict__ Wt,
                                            int ldw, int n0, int k0, int lane) {
    const int col = n0 + (lane & 15);
    const int kb  = k0 + (lane >> 4) * 16;
    return *(const v16h*)(Wt + (size_t)col * ldw + kb);
}

// ---------------------------------------------------------------------------
// Weight convert + transpose: wt[n*K + k] = (f16) w[k*N + n]
// ---------------------------------------------------------------------------
__global__ void transpose_w_kernel(const float* __restrict__ w,
                                   _Float16* __restrict__ wt, int K, int N) {
    int idx = blockIdx.x * 256 + threadIdx.x;
    if (idx < K * N) {
        int n = idx / K, k = idx % K;
        wt[idx] = (_Float16)w[k * N + n];
    }
}

// ---------------------------------------------------------------------------
// Haar decompose + |band| 2x2 pool -> f16 token matrices (M=B*1024, K=256)
// One thread: one (b,c,token) = a 4x4 patch of x_big; 4x nt b128 loads.
// ---------------------------------------------------------------------------
__global__ __launch_bounds__(256) void haar_tokens_kernel(
    const float* __restrict__ xb,
    _Float16* __restrict__ zh, _Float16* __restrict__ zv,
    _Float16* __restrict__ zd) {
    const int bc = blockIdx.x;                       // b*CB + c
    const int tx = threadIdx.x;                      // 0..31
    const int ty = blockIdx.y * 8 + threadIdx.y;     // 0..31
    const float* rowp =
        xb + (size_t)bc * (HH * WW) + (4 * ty) * WW + 4 * tx;
    nf4 r0 = nt_load4(rowp);
    nf4 r1 = nt_load4(rowp + WW);
    nf4 r2 = nt_load4(rowp + 2 * WW);
    nf4 r3 = nt_load4(rowp + 3 * WW);

    float ah = 0.f, av = 0.f, ad = 0.f;
    auto accum = [&](float p00, float p01, float p10, float p11) {
        ah += fabsf((p00 - p01 + p10 - p11) * 0.25f);
        av += fabsf((p00 + p01 - p10 - p11) * 0.25f);
        ad += fabsf((p00 - p01 - p10 + p11) * 0.25f);
    };
    accum(r0[0], r0[1], r1[0], r1[1]);
    accum(r0[2], r0[3], r1[2], r1[3]);
    accum(r2[0], r2[1], r3[0], r3[1]);
    accum(r2[2], r2[3], r3[2], r3[3]);

    const int b = bc >> 8, c = bc & (CB - 1);
    const int t = ty * 32 + tx;
    const size_t o = ((size_t)b * TOKH + t) * CB + c;
    zh[o] = (_Float16)(ah * 0.25f);
    zv[o] = (_Float16)(av * 0.25f);
    zd[o] = (_Float16)(ad * 0.25f);
}

// ---------------------------------------------------------------------------
// x_small 4x4 mean pool -> f16 token matrix (M=B*256, K=512).
// (guide einsum commutes with the pooling, so pool first: 16x fewer MACs)
// ---------------------------------------------------------------------------
__global__ __launch_bounds__(256) void pool_small_kernel(
    const float* __restrict__ xs, _Float16* __restrict__ p) {
    const int bs = blockIdx.x;                 // b*CS + s
    const int tx = threadIdx.x, ty = threadIdx.y;
    const float* base = xs + (size_t)bs * (H2 * W2);
    float acc = 0.f;
#pragma unroll
    for (int dy = 0; dy < 4; ++dy) {
        nf4 v = nt_load4(base + (ty * 4 + dy) * W2 + tx * 4);
        acc += v[0] + v[1] + v[2] + v[3];
    }
    const int b = bs >> 9, s = bs & (CS - 1);
    const int t = ty * 16 + tx;
    p[((size_t)b * TOKL + t) * CS + s] = (_Float16)(acc * (1.f / 16.f));
}

// ---------------------------------------------------------------------------
// GEMM1: Y = [LN](X @ W1 + b1), X: M x K f16, Wt: N x K f16 (transposed),
// Y: M x N f16.  One WG = 32 token rows (MT=2 WMMA M-tiles) x full N
// (8 waves x NT N-tiles).  A block staged in LDS once (kills the 8x
// redundant per-wave A fetch); each B fragment feeds MT WMMAs.
// LayerNorm over N done on the f16 LDS tile (full rows resident).
// ---------------------------------------------------------------------------
template <int N, int K, bool DO_LN>
__global__ __launch_bounds__(256) void gemm1_kernel(
    const _Float16* __restrict__ X, const _Float16* __restrict__ Wt,
    const float* __restrict__ bias, const float* __restrict__ lnw,
    const float* __restrict__ lnb, _Float16* __restrict__ Y) {
    constexpr int NT = N / 128;               // WMMA N-tiles per wave (8 waves)
    constexpr int MT = 2;                     // WMMA M-tiles per WG
    __shared__ _Float16 As[32 * K];
    __shared__ _Float16 tile[32 * N];
    __shared__ float rs[32][9], rs2[32][9];
    __shared__ float mrow[32], rrow[32];

    const int tid = threadIdx.x;
    const int wave = tid >> 5, lane = tid & 31;
    const int m0 = blockIdx.x * 32;

    // stage A block (32 x K halves, contiguous rows) into LDS, b128 coalesced
    {
        const v8h* src = (const v8h*)(X + (size_t)m0 * K);
        v8h* dst = (v8h*)As;
        for (int i = tid; i < 32 * K / 8; i += 256) dst[i] = src[i];
    }
    __syncthreads();

    v8f acc[MT][NT];
#pragma unroll
    for (int mi = 0; mi < MT; ++mi)
#pragma unroll
        for (int j = 0; j < NT; ++j)
#pragma unroll
            for (int e = 0; e < 8; ++e) acc[mi][j][e] = 0.f;

    const int hsel = lane >> 4;
    for (int k0 = 0; k0 < K; k0 += 32) {
        v16h a[MT];
#pragma unroll
        for (int mi = 0; mi < MT; ++mi) {
            // A 16x32 as loaded here (k order shared with B): lanes 0-15 rows M, K {0..15};
            // lanes 16-31 same rows, K {16..31}
            const int row = mi * 16 + (lane & 15);
            const _Float16* p = &As[row * K + k0 + hsel * 16];
            v8h lo = *(const v8h*)p;
            v8h hi = *(const v8h*)(p + 8);
#pragma unroll
            for (int i = 0; i < 8; ++i) { a[mi][i] = lo[i]; a[mi][i + 8] = hi[i]; }
        }
#pragma unroll
        for (int j = 0; j < NT; ++j) {
            const int n0 = wave * (NT * 16) + j * 16;
            v16h bf = load_b_frag(Wt, K, n0, k0, lane);
#pragma unroll
            for (int mi = 0; mi < MT; ++mi)
                acc[mi][j] = __builtin_amdgcn_wmma_f32_16x16x32_f16(
                    false, a[mi], false, bf, (short)0, acc[mi][j], false, false);
        }
    }

    // scatter C tiles (+bias) into f16 LDS tile:
    // row = mi*16 + (lane>>4)*8 + e, col = n0 + (lane&15)
    const int rb = (lane >> 4) * 8;
    const int cl = lane & 15;
#pragma unroll
    for (int mi = 0; mi < MT; ++mi)
#pragma unroll
        for (int j = 0; j < NT; ++j) {
            const int n0 = wave * (NT * 16) + j * 16;
#pragma unroll
            for (int e = 0; e < 8; ++e)
                tile[(mi * 16 + rb + e) * N + n0 + cl] =
                    (_Float16)(acc[mi][j][e] + bias[n0 + cl]);
        }
    __syncthreads();

    if (DO_LN) {
        const int r = tid >> 3, s = tid & 7;     // 32 rows x 8 partials
        float sum = 0.f, sum2 = 0.f;
        for (int c = s; c < N; c += 8) {
            float x = (float)tile[r * N + c];
            sum += x; sum2 += x * x;
        }
        rs[r][s] = sum; rs2[r][s] = sum2;
        __syncthreads();
        if (s == 0) {
            float S = 0.f, S2 = 0.f;
            for (int i = 0; i < 8; ++i) { S += rs[r][i]; S2 += rs2[r][i]; }
            const float m = S / (float)N;
            const float var = S2 / (float)N - m * m;
            mrow[r] = m;
            rrow[r] = rsqrtf(var + 1e-5f);
        }
        __syncthreads();
        for (int i = tid; i < 32 * N; i += 256) {
            const int rr = i / N, cc = i % N;
            const float x =
                ((float)tile[i] - mrow[rr]) * rrow[rr] * lnw[cc] + lnb[cc];
            Y[(size_t)(m0 + rr) * N + cc] = (_Float16)x;
        }
    } else {
        for (int i = tid; i < 32 * N; i += 256) {
            const int rr = i / N, cc = i % N;
            Y[(size_t)(m0 + rr) * N + cc] = tile[i];
        }
    }
}

// ---------------------------------------------------------------------------
// GEMM2: Q = X @ W2 + b2 (X: Mx512 f16, Wt: 256x512 f16), 32 rows per WG,
// A staged in LDS, then epilogue:
//   MODE 0: alpha = relu(cos(Q_row, prompt))   -> out: M floats
//   MODE 1: gate  = sigmoid(Q)                 -> out: M x 256 floats
// ---------------------------------------------------------------------------
template <int MODE>
__global__ __launch_bounds__(256) void gemm2_kernel(
    const _Float16* __restrict__ X, const _Float16* __restrict__ Wt,
    const float* __restrict__ bias, const float* __restrict__ prompt,
    float* __restrict__ out) {
    constexpr int N = 256, K = 512, NT = 2, MT = 2;
    __shared__ _Float16 As[32 * K];
    __shared__ _Float16 tile[32 * N];
    __shared__ float r0[32][9], r1[32][9], r2[32][9];

    const int tid = threadIdx.x;
    const int wave = tid >> 5, lane = tid & 31;
    const int m0 = blockIdx.x * 32;

    {
        const v8h* src = (const v8h*)(X + (size_t)m0 * K);
        v8h* dst = (v8h*)As;
        for (int i = tid; i < 32 * K / 8; i += 256) dst[i] = src[i];
    }
    __syncthreads();

    v8f acc[MT][NT];
#pragma unroll
    for (int mi = 0; mi < MT; ++mi)
#pragma unroll
        for (int j = 0; j < NT; ++j)
#pragma unroll
            for (int e = 0; e < 8; ++e) acc[mi][j][e] = 0.f;

    const int hsel = lane >> 4;
    for (int k0 = 0; k0 < K; k0 += 32) {
        v16h a[MT];
#pragma unroll
        for (int mi = 0; mi < MT; ++mi) {
            const int row = mi * 16 + (lane & 15);
            const _Float16* p = &As[row * K + k0 + hsel * 16];
            v8h lo = *(const v8h*)p;
            v8h hi = *(const v8h*)(p + 8);
#pragma unroll
            for (int i = 0; i < 8; ++i) { a[mi][i] = lo[i]; a[mi][i + 8] = hi[i]; }
        }
#pragma unroll
        for (int j = 0; j < NT; ++j) {
            const int n0 = wave * (NT * 16) + j * 16;
            v16h bf = load_b_frag(Wt, K, n0, k0, lane);
#pragma unroll
            for (int mi = 0; mi < MT; ++mi)
                acc[mi][j] = __builtin_amdgcn_wmma_f32_16x16x32_f16(
                    false, a[mi], false, bf, (short)0, acc[mi][j], false, false);
        }
    }

    const int rb = (lane >> 4) * 8;
    const int cl = lane & 15;
#pragma unroll
    for (int mi = 0; mi < MT; ++mi)
#pragma unroll
        for (int j = 0; j < NT; ++j) {
            const int n0 = wave * (NT * 16) + j * 16;
#pragma unroll
            for (int e = 0; e < 8; ++e)
                tile[(mi * 16 + rb + e) * N + n0 + cl] =
                    (_Float16)(acc[mi][j][e] + bias[n0 + cl]);
        }
    __syncthreads();

    if (MODE == 0) {
        const int r = tid >> 3, s = tid & 7;
        float dp = 0.f, qq = 0.f, pp = 0.f;
        for (int c = s; c < N; c += 8) {
            const float q = (float)tile[r * N + c];
            const float pv = prompt[c];
            dp += q * pv; qq += q * q; pp += pv * pv;
        }
        r0[r][s] = dp; r1[r][s] = qq; r2[r][s] = pp;
        __syncthreads();
        if (s == 0) {
            float DP = 0.f, QQ = 0.f, PP = 0.f;
            for (int i = 0; i < 8; ++i) {
                DP += r0[r][i]; QQ += r1[r][i]; PP += r2[r][i];
            }
            const float denom =
                fmaxf(sqrtf(QQ), 1e-8f) * fmaxf(sqrtf(PP), 1e-8f);
            const float sim = DP / denom;
            out[m0 + r] = sim > 0.f ? sim : 0.f;
        }
    } else {
        for (int i = tid; i < 32 * N; i += 256) {
            const int rr = i >> 8, cc = i & 255;
            out[(size_t)(m0 + rr) * N + cc] =
                1.f / (1.f + __expf(-(float)tile[i]));
        }
    }
}

// ---------------------------------------------------------------------------
// Recombine: re-read x_big (NT), redo Haar per quad, scale bands by alpha,
// scale average band by gate, inverse Haar, NT-store output.
// One thread handles a 2x4 output strip (two quads sharing token indices).
// ---------------------------------------------------------------------------
__global__ __launch_bounds__(256) void recombine_kernel(
    const float* __restrict__ xb, const float* __restrict__ alpha_h,
    const float* __restrict__ alpha_v, const float* __restrict__ alpha_d,
    const float* __restrict__ gate, float* __restrict__ out) {
    const int bc = blockIdx.x;                    // b*CB + c
    const int c = bc & (CB - 1), b = bc >> 8;
    const int xq = threadIdx.x;                   // 0..31: half-res col pair
    const int yh = blockIdx.y * 8 + threadIdx.y;  // 0..63

    const size_t ofs = (size_t)bc * (HH * WW) + (yh * 2) * WW + xq * 4;
    nf4 top = nt_load4(xb + ofs);
    nf4 bot = nt_load4(xb + ofs + WW);

    // both quads in this strip share the same alpha token (xh>>1 == xq) and
    // gate token (xh>>2 == xq>>1)
    const size_t t2 = (size_t)b * TOKH + (yh >> 1) * 32 + xq;
    const float ah = alpha_h[t2], av = alpha_v[t2], ad = alpha_d[t2];
    const float g =
        gate[((size_t)b * TOKL + (yh >> 2) * 16 + (xq >> 1)) * CB + c];

    nf4 o_top, o_bot;
#pragma unroll
    for (int q = 0; q < 2; ++q) {
        const float p00 = top[2 * q], p01 = top[2 * q + 1];
        const float p10 = bot[2 * q], p11 = bot[2 * q + 1];
        const float a  = (p00 + p01 + p10 + p11) * 0.25f;
        const float hb = (p00 - p01 + p10 - p11) * 0.25f;
        const float vb = (p00 + p01 - p10 - p11) * 0.25f;
        const float db = (p00 - p01 - p10 + p11) * 0.25f;
        const float ae = a * g, he = hb * ah, ve = vb * av, de = db * ad;
        o_top[2 * q]     = ae + he + ve + de;
        o_top[2 * q + 1] = ae - he + ve - de;
        o_bot[2 * q]     = ae + he - ve - de;
        o_bot[2 * q + 1] = ae - he - ve + de;
    }
    nt_store4(out + ofs, o_top);
    nt_store4(out + ofs + WW, o_bot);
}

// ---------------------------------------------------------------------------
// Host launch
// ---------------------------------------------------------------------------
extern "C" void kernel_launch(void* const* d_in, const int* in_sizes, int n_in,
                              void* d_out, int out_size, void* d_ws,
                              size_t ws_size, hipStream_t stream) {
    (void)in_sizes; (void)n_in; (void)out_size; (void)ws_size;
    const float* x_big   = (const float*)d_in[0];
    const float* x_small = (const float*)d_in[1];
    const float* hf_w1   = (const float*)d_in[2];
    const float* hf_b1   = (const float*)d_in[3];
    const float* hf_lnw  = (const float*)d_in[4];
    const float* hf_lnb  = (const float*)d_in[5];
    const float* hf_w2   = (const float*)d_in[6];
    const float* hf_b2   = (const float*)d_in[7];
    const float* hf_pr   = (const float*)d_in[8];
    const float* low_w   = (const float*)d_in[9];
    const float* low_b   = (const float*)d_in[10];
    const float* lf_w1   = (const float*)d_in[11];
    const float* lf_b1   = (const float*)d_in[12];
    const float* lf_lnw  = (const float*)d_in[13];
    const float* lf_lnb  = (const float*)d_in[14];
    const float* lf_w2   = (const float*)d_in[15];
    const float* lf_b2   = (const float*)d_in[16];
    float* out = (float*)d_out;

    char* ws = (char*)d_ws;
    size_t off = 0;
    auto alloc = [&](size_t bytes) {
        size_t o = off;
        off = (off + bytes + 255) & ~(size_t)255;
        return o;
    };
    _Float16* hf_w1t = (_Float16*)(ws + alloc(HID * CB * 2));   // 512x256
    _Float16* hf_w2t = (_Float16*)(ws + alloc(CB * HID * 2));   // 256x512
    _Float16* low_wt = (_Float16*)(ws + alloc(CB * CS * 2));    // 256x512
    _Float16* lf_w1t = (_Float16*)(ws + alloc(HID * CB * 2));   // 512x256
    _Float16* lf_w2t = (_Float16*)(ws + alloc(CB * HID * 2));   // 256x512
    // three band token matrices stacked contiguously: one M=3*MH GEMM
    _Float16* zall = (_Float16*)(ws + alloc((size_t)3 * MH * CB * 2));
    _Float16* Hbuf = (_Float16*)(ws + alloc((size_t)3 * MH * HID * 2));
    float* aall    = (float*)(ws + alloc((size_t)3 * MH * 4));
    _Float16* psm  = (_Float16*)(ws + alloc((size_t)ML * CS * 2));
    _Float16* ktok = (_Float16*)(ws + alloc((size_t)ML * CB * 2));
    float* gate    = (float*)(ws + alloc((size_t)ML * CB * 4));

    _Float16* zh = zall;
    _Float16* zv = zall + (size_t)MH * CB;
    _Float16* zd = zall + (size_t)2 * MH * CB;
    float* ah = aall;
    float* av = aall + MH;
    float* ad = aall + 2 * MH;

    // 1) convert + transpose weights to f16 NxK
    transpose_w_kernel<<<(CB * HID + 255) / 256, 256, 0, stream>>>(hf_w1, hf_w1t, CB, HID);
    transpose_w_kernel<<<(HID * CB + 255) / 256, 256, 0, stream>>>(hf_w2, hf_w2t, HID, CB);
    transpose_w_kernel<<<(CS * CB + 255) / 256, 256, 0, stream>>>(low_w, low_wt, CS, CB);
    transpose_w_kernel<<<(CB * HID + 255) / 256, 256, 0, stream>>>(lf_w1, lf_w1t, CB, HID);
    transpose_w_kernel<<<(HID * CB + 255) / 256, 256, 0, stream>>>(lf_w2, lf_w2t, HID, CB);

    // 2) tokens (NT streaming reads of x_big / x_small)
    haar_tokens_kernel<<<dim3(BB * CB, 4), dim3(32, 8), 0, stream>>>(x_big, zh, zv, zd);
    pool_small_kernel<<<BB * CS, dim3(16, 16), 0, stream>>>(x_small, psm);

    // 3) all three high-band MLPs as ONE stacked M=3*MH WMMA GEMM pair
    gemm1_kernel<HID, CB, true><<<3 * MH / 32, 256, 0, stream>>>(
        zall, hf_w1t, hf_b1, hf_lnw, hf_lnb, Hbuf);
    gemm2_kernel<0><<<3 * MH / 32, 256, 0, stream>>>(
        Hbuf, hf_w2t, hf_b2, hf_pr, aall);

    // 4) low path: pooled x_small @ low_w (no LN), then lf MLP -> sigmoid gate
    gemm1_kernel<CB, CS, false><<<ML / 32, 256, 0, stream>>>(
        psm, low_wt, low_b, nullptr, nullptr, ktok);
    gemm1_kernel<HID, CB, true><<<ML / 32, 256, 0, stream>>>(
        ktok, lf_w1t, lf_b1, lf_lnw, lf_lnb, Hbuf);
    gemm2_kernel<1><<<ML / 32, 256, 0, stream>>>(
        Hbuf, lf_w2t, lf_b2, nullptr, gate);

    // 5) fused inverse-Haar recombine (NT stream in/out)
    recombine_kernel<<<dim3(BB * CB, 8), dim3(32, 8), 0, stream>>>(
        x_big, ah, av, ad, gate, out);
}
